// CrossAttention3D_85856396247528
// MI455X (gfx1250) — matrix operand order repaired and checked by reading, not compiled, not run on hardware
//
#include <hip/hip_runtime.h>

// ---------------------------------------------------------------------------
// CrossAttention (B=4, C=512, H=W=64 -> S=4096), single head.
// QKV proj (bf16 WMMA GEMM) -> flash attention (bf16 WMMA, online softmax,
// f32 accum) -> output proj (bf16 WMMA GEMM, f32 store).
// All matrix math via v_wmma_f32_16x16x32_bf16 (gfx1250, wave32).
// ---------------------------------------------------------------------------

#define S_LEN 4096
#define CDIM  512
#define NBATCH 4

typedef __attribute__((ext_vector_type(16))) __bf16 v16bf;
typedef __attribute__((ext_vector_type(8)))  __bf16 v8bf;
typedef __attribute__((ext_vector_type(4)))  __bf16 v4bf;
typedef __attribute__((ext_vector_type(8)))  float  v8f;
typedef __attribute__((ext_vector_type(4)))  float  v4f;
typedef __attribute__((ext_vector_type(4)))  int    v4i;

// ---- optional CDNA5 async global->LDS path (guarded; sync fallback) -------
#if defined(__has_builtin)
#if __has_builtin(__builtin_amdgcn_global_load_async_to_lds_b128) && \
    __has_builtin(__builtin_amdgcn_s_wait_asynccnt)
#define USE_ASYNC_LDS 1
#endif
#endif
#ifndef USE_ASYNC_LDS
#define USE_ASYNC_LDS 0
#endif

__device__ __forceinline__ void async_copy_b128(void* lds, const void* g) {
#if USE_ASYNC_LDS
  typedef __attribute__((address_space(3))) v4i* lds_v4i_p;
  __builtin_amdgcn_global_load_async_to_lds_b128(
      (v4i*)g, (lds_v4i_p)(v4i*)lds, 0, 0);
#else
  *(v8bf*)lds = *(const v8bf*)g;
#endif
}

__device__ __forceinline__ void async_wait() {
#if USE_ASYNC_LDS
  __builtin_amdgcn_s_wait_asynccnt(0);
#endif
}

// ---- WMMA wrapper with compile-time reuse (OPSEL RA/RB) hints -------------
template <bool RA, bool RB>
__device__ __forceinline__ v8f wmma_bf16(v16bf a, v16bf b, v8f c) {
  // 8 args: (neg_a, A, neg_b, B, c_mod, C, reuse_a, reuse_b)
  return __builtin_amdgcn_wmma_f32_16x16x32_bf16(false, a, false, b, (short)0, c,
                                                 RA, RB);
}

// A-matrix fragment (16x32 bf16, MxK). ISA layout: lane holds row m = lane&15;
// lanes 0-15 hold K pairs {0..7,16..23}, lanes 16-31 hold {8..15,24..31}.
// Row-major source (stride ldk elems, 16B-aligned rows): two 16B reads/lane.
__device__ __forceinline__ v16bf load_frag_a(const __bf16* __restrict__ base,
                                             int ldk, int kbase) {
  const int lane  = threadIdx.x & 31;
  const int m     = lane & 15;
  const int lhalf = lane >> 4;
  const __bf16* p = base + (size_t)m * ldk + kbase + lhalf * 8;
  v8bf lo = *(const v8bf*)(p);
  v8bf hi = *(const v8bf*)(p + 16);
  v16bf f;
#pragma unroll
  for (int i = 0; i < 8; ++i) { f[i] = lo[i]; f[8 + i] = hi[i]; }
  return f;
}

// B-matrix fragment (32x16 bf16, KxN). ISA layout: lane holds column
// n = lane&15; lanes 0-15 hold K {0..7,16..23}, lanes 16-31 hold {8..15,24..31}.
// Source stored "n-major" (each n is a row of contiguous K, stride ldk).
__device__ __forceinline__ v16bf load_frag_b(const __bf16* __restrict__ base,
                                             int ldk, int kbase) {
  const int lane  = threadIdx.x & 31;
  const int n     = lane & 15;
  const int lhalf = lane >> 4;
  const __bf16* p = base + (size_t)n * ldk + kbase + lhalf * 8;
  v8bf lo = *(const v8bf*)(p);
  v8bf hi = *(const v8bf*)(p + 16);
  v16bf f;
#pragma unroll
  for (int i = 0; i < 8; ++i) { f[i] = lo[i]; f[8 + i] = hi[i]; }
  return f;
}

// ---------------------------------------------------------------------------
// Projection GEMM: Y[s,o] = sum_c A[s,c] * W[o,c] + bias[o]
// MODE 0: Xf fp32 (C,S) in  -> Ybf bf16 (S,C)  [Q,K]
// MODE 1: Xf fp32 (C,S) in  -> Ybf bf16 (C,S)  [V]
// MODE 2: Xbf bf16 (S,C) in -> Yf32 fp32 (O,S) [final]
// Block: 256 threads / 8 waves; tile 64(s) x 64(o); K-step 32.
// ---------------------------------------------------------------------------
template <int MODE>
__global__ __launch_bounds__(256) void proj_gemm_kernel(
    const float* __restrict__ Xf, const __bf16* __restrict__ Xbf,
    const float* __restrict__ W, const float* __restrict__ bias,
    __bf16* __restrict__ Ybf, float* __restrict__ Yf32) {
  __shared__ __bf16 lds_A[64][32];  // [s_local][c_local]
  __shared__ __bf16 lds_W[64][32];  // [o_local][c_local]

  const int s0 = blockIdx.x * 64;
  const int o0 = blockIdx.y * 64;
  const int bb = blockIdx.z;

  const int tid  = threadIdx.x;
  const int lane = tid & 31;
  const int w    = tid >> 5;
  const int wm   = w >> 1;  // 0..3  (16-row slice)
  const int wn   = w & 1;   // 0..1  (32-col slice)

  const size_t batch_off = (size_t)bb * CDIM * S_LEN;

  v8f acc[2];
  {
    const int nl = lane & 15;
    const float b0 = bias[o0 + wn * 32 + nl];
    const float b1 = bias[o0 + wn * 32 + 16 + nl];
#pragma unroll
    for (int r = 0; r < 8; ++r) { acc[0][r] = b0; acc[1][r] = b1; }
  }

  for (int kc = 0; kc < CDIM; kc += 32) {
    __syncthreads();
    // ---- stage activation tile (transposed to (s,c), bf16), vectorized ----
    if (MODE == 2) {
      // bf16 (S,C): rows already contiguous -> one 16B async copy per thread.
      const int row = tid >> 2, ch = (tid & 3) * 8;
      async_copy_b128(&lds_A[row][ch],
                      &Xbf[batch_off + (size_t)(s0 + row) * CDIM + kc + ch]);
      if (kc + 32 < CDIM)
        __builtin_prefetch(
            &Xbf[batch_off + (size_t)(s0 + row) * CDIM + kc + 32 + ch], 0, 3);
    } else {
      // fp32 (C,S): float4 along s (coalesced), transpose into LDS.
#pragma unroll
      for (int rep = 0; rep < 2; ++rep) {
        const int e  = tid + rep * 256;  // 0..511 float4 units
        const int c  = e >> 4;           // 0..31
        const int s4 = (e & 15) * 4;
        const v4f xv =
            *(const v4f*)&Xf[batch_off + (size_t)(kc + c) * S_LEN + s0 + s4];
#pragma unroll
        for (int i = 0; i < 4; ++i) lds_A[s4 + i][c] = (__bf16)xv[i];
        if (kc + 32 < CDIM)
          __builtin_prefetch(
              &Xf[batch_off + (size_t)(kc + 32 + c) * S_LEN + s0 + s4], 0, 3);
      }
    }
    // ---- stage weight k-tile as bf16 (row-major (o,c)), vectorized --------
#pragma unroll
    for (int rep = 0; rep < 2; ++rep) {
      const int e  = tid + rep * 256;  // 0..511 float4 units
      const int o  = e >> 3;           // 0..63
      const int c4 = (e & 7) * 4;
      const v4f wv = *(const v4f*)&W[(size_t)(o0 + o) * CDIM + kc + c4];
      v4bf pk;
#pragma unroll
      for (int i = 0; i < 4; ++i) pk[i] = (__bf16)wv[i];
      *(v4bf*)&lds_W[o][c4] = pk;
    }
    if (MODE == 2) async_wait();
    __syncthreads();

    const v16bf a  = load_frag_a(&lds_A[wm * 16][0], 32, 0);
    const v16bf b0 = load_frag_b(&lds_W[wn * 32][0], 32, 0);
    const v16bf b1 = load_frag_b(&lds_W[wn * 32 + 16][0], 32, 0);
    acc[0] = wmma_bf16<false, false>(a, b0, acc[0]);
    acc[1] = wmma_bf16<true, false>(a, b1, acc[1]);  // same A as previous WMMA
  }

  // Epilogue. C/D layout: VGPR r -> M = r + 8*(lane>=16), N = lane&15.
  const int lhalf = lane >> 4, nl = lane & 15;
#pragma unroll
  for (int n = 0; n < 2; ++n) {
#pragma unroll
    for (int r = 0; r < 8; ++r) {
      const int s = s0 + wm * 16 + r + lhalf * 8;
      const int o = o0 + wn * 32 + n * 16 + nl;
      const float v = acc[n][r];
      if (MODE == 2) {
        Yf32[batch_off + (size_t)o * S_LEN + s] = v;         // (B,O,S) fp32
      } else if (MODE == 1) {
        Ybf[batch_off + (size_t)o * S_LEN + s] = (__bf16)v;  // V: (B,C,S)
      } else {
        Ybf[batch_off + (size_t)s * CDIM + o] = (__bf16)v;   // Q/K: (B,S,C)
      }
    }
  }
}

// ---------------------------------------------------------------------------
// Flash attention. Q,K: (B,S,C) bf16. V: (B,C,S) bf16. O: (B,S,C) bf16.
// Block: 256 threads / 8 waves; 32 query rows per block; key tiles of 64.
// Wave (wm, wc): wm in {0,1} = 16-row slice, wc in {0..3} = 128-col C slice.
// ---------------------------------------------------------------------------
__global__ __launch_bounds__(256) void flash_attn_kernel(
    const __bf16* __restrict__ Q, const __bf16* __restrict__ Kmat,
    const __bf16* __restrict__ V, __bf16* __restrict__ O) {
  __shared__ float  sim[32][68];   // stride 68: 16B-aligned rows + bank rotate
  __shared__ __bf16 Pt[32][64];    // exp(sim - m) in A-fragment layout
  __shared__ float  mrow[32], lrow[32], crow[32];

  const int q0  = blockIdx.x * 32;
  const int bb  = blockIdx.y;
  const int tid = threadIdx.x, lane = tid & 31, w = tid >> 5;
  const int wm = w >> 2, wc = w & 3;
  const float scale = 0.04419417382415922f;  // 512^-0.5

  const __bf16* Qb = Q + (size_t)bb * S_LEN * CDIM;
  const __bf16* Kb = Kmat + (size_t)bb * S_LEN * CDIM;
  const __bf16* Vb = V + (size_t)bb * CDIM * S_LEN;

  // Q fragments for this wave's 16 rows, all 512 K: resident in VGPRs,
  // reused across all 64 key tiles.
  v16bf qf[16];
#pragma unroll
  for (int kk = 0; kk < 16; ++kk)
    qf[kk] = load_frag_a(Qb + (size_t)(q0 + wm * 16) * CDIM, CDIM, kk * 32);

  v8f acc[8];
#pragma unroll
  for (int n = 0; n < 8; ++n)
#pragma unroll
    for (int r = 0; r < 8; ++r) acc[n][r] = 0.0f;

  if (tid < 32) { mrow[tid] = -1e30f; lrow[tid] = 0.0f; }
  __syncthreads();

  for (int j0 = 0; j0 < S_LEN; j0 += 64) {
    // ---- phase 1: each wave computes one 16x16 sim tile (rows wm, cols wc)
    // Software-pipelined: groups of {4 B-fragment loads -> 4 WMMAs} so later
    // groups' loads overlap in-flight WMMAs (partial s_wait_loadcnt).
    v8f sa;
#pragma unroll
    for (int r = 0; r < 8; ++r) sa[r] = 0.0f;
    {
      const __bf16* kb = Kb + (size_t)(j0 + wc * 16) * CDIM;
#pragma unroll
      for (int g = 0; g < 4; ++g) {
        v16bf bf[4];
#pragma unroll
        for (int i = 0; i < 4; ++i)
          bf[i] = load_frag_b(kb, CDIM, (g * 4 + i) * 32);
#pragma unroll
        for (int i = 0; i < 4; ++i)
          sa = wmma_bf16<false, false>(qf[g * 4 + i], bf[i], sa);
      }
    }
    {
      const int lhalf = lane >> 4, nl = lane & 15;
#pragma unroll
      for (int r = 0; r < 8; ++r)
        sim[wm * 16 + r + lhalf * 8][wc * 16 + nl] = sa[r] * scale;
    }
    if (j0 + 64 < S_LEN) {  // hint next K/V tiles into cache
      __builtin_prefetch(Kb + (size_t)(j0 + 64 + wc * 16) * CDIM, 0, 3);
      __builtin_prefetch(Vb + (size_t)(wc * 128) * S_LEN + j0 + 64, 0, 3);
    }
    __syncthreads();

    // ---- phase 2a: online softmax, 8 lanes per row (wave-local groups) ----
    {
      const int row = tid >> 3;        // 0..31
      const int seg = (tid & 7) * 8;   // 8 columns per lane
      const v4f s0v = *(const v4f*)&sim[row][seg];
      const v4f s1v = *(const v4f*)&sim[row][seg + 4];
      float mx = s0v[0];
#pragma unroll
      for (int i = 1; i < 4; ++i) mx = fmaxf(mx, s0v[i]);
#pragma unroll
      for (int i = 0; i < 4; ++i) mx = fmaxf(mx, s1v[i]);
#pragma unroll
      for (int off = 1; off < 8; off <<= 1) mx = fmaxf(mx, __shfl_xor(mx, off, 32));
      const float mo = mrow[row];
      mx = fmaxf(mx, mo);
      float sum = 0.0f;
      v8bf pv;
#pragma unroll
      for (int i = 0; i < 4; ++i) {
        const float p = __expf(s0v[i] - mx);
        pv[i] = (__bf16)p;
        sum += p;
      }
#pragma unroll
      for (int i = 0; i < 4; ++i) {
        const float p = __expf(s1v[i] - mx);
        pv[4 + i] = (__bf16)p;
        sum += p;
      }
      *(v8bf*)&Pt[row][seg] = pv;
#pragma unroll
      for (int off = 1; off < 8; off <<= 1) sum += __shfl_xor(sum, off, 32);
      if ((tid & 7) == 0) {  // one writer per row; row group is wave-local
        const float corr = __expf(mo - mx);
        mrow[row] = mx;
        lrow[row] = lrow[row] * corr + sum;
        crow[row] = corr;
      }
    }
    __syncthreads();

    // ---- phase 2b: rescale accumulator, then acc += P x V -----------------
    const int lhalf = lane >> 4;
    float cf[8];
#pragma unroll
    for (int r = 0; r < 8; ++r) cf[r] = crow[wm * 16 + r + lhalf * 8];
#pragma unroll
    for (int n = 0; n < 8; ++n)
#pragma unroll
      for (int r = 0; r < 8; ++r) acc[n][r] *= cf[r];

    v16bf pa[2];
    pa[0] = load_frag_a(&Pt[wm * 16][0], 64, 0);
    pa[1] = load_frag_a(&Pt[wm * 16][0], 64, 32);
#pragma unroll
    for (int kk = 0; kk < 2; ++kk) {
      // Pipelined groups of 4; consecutive WMMAs share A (pa[kk]) -> reuse_a.
      v16bf bv[4];
#pragma unroll
      for (int i = 0; i < 4; ++i)
        bv[i] = load_frag_b(Vb + (size_t)(wc * 128 + i * 16) * S_LEN + j0,
                            S_LEN, kk * 32);
      acc[0] = wmma_bf16<false, false>(pa[kk], bv[0], acc[0]);
      acc[1] = wmma_bf16<true, false>(pa[kk], bv[1], acc[1]);
      acc[2] = wmma_bf16<true, false>(pa[kk], bv[2], acc[2]);
      acc[3] = wmma_bf16<true, false>(pa[kk], bv[3], acc[3]);
#pragma unroll
      for (int i = 0; i < 4; ++i)
        bv[i] = load_frag_b(Vb + (size_t)(wc * 128 + (4 + i) * 16) * S_LEN + j0,
                            S_LEN, kk * 32);
      acc[4] = wmma_bf16<true, false>(pa[kk], bv[0], acc[4]);
      acc[5] = wmma_bf16<true, false>(pa[kk], bv[1], acc[5]);
      acc[6] = wmma_bf16<true, false>(pa[kk], bv[2], acc[6]);
      acc[7] = wmma_bf16<true, false>(pa[kk], bv[3], acc[7]);
    }
    __syncthreads();
  }

  // ---- epilogue: divide by row sum, store (B,S,C) bf16 --------------------
  const int lhalf = lane >> 4, nl = lane & 15;
  float il[8];
#pragma unroll
  for (int r = 0; r < 8; ++r) il[r] = 1.0f / lrow[wm * 16 + r + lhalf * 8];
#pragma unroll
  for (int n = 0; n < 8; ++n)
#pragma unroll
    for (int r = 0; r < 8; ++r) {
      const int s = q0 + wm * 16 + r + lhalf * 8;
      const int c = wc * 128 + n * 16 + nl;
      O[(size_t)bb * S_LEN * CDIM + (size_t)s * CDIM + c] =
          (__bf16)(acc[n][r] * il[r]);
    }
}

// ---------------------------------------------------------------------------
extern "C" void kernel_launch(void* const* d_in, const int* in_sizes, int n_in,
                              void* d_out, int out_size, void* d_ws, size_t ws_size,
                              hipStream_t stream) {
  const float* q  = (const float*)d_in[0];
  const float* Wq = (const float*)d_in[1];
  const float* bq = (const float*)d_in[2];
  const float* Wk = (const float*)d_in[3];
  const float* bk = (const float*)d_in[4];
  const float* Wv = (const float*)d_in[5];
  const float* bv = (const float*)d_in[6];
  const float* Wo = (const float*)d_in[7];
  const float* bo = (const float*)d_in[8];
  float* out = (float*)d_out;

  const size_t elems = (size_t)NBATCH * CDIM * S_LEN;  // 8.39M
  __bf16* Qb = (__bf16*)d_ws;        // (B,S,C)
  __bf16* Kb = Qb + elems;           // (B,S,C)
  __bf16* Vb = Kb + elems;           // (B,C,S)
  __bf16* Ob = Vb + elems;           // (B,S,C)

  const dim3 gp(S_LEN / 64, CDIM / 64, NBATCH);
  proj_gemm_kernel<0><<<gp, 256, 0, stream>>>(q, nullptr, Wq, bq, Qb, nullptr);
  proj_gemm_kernel<0><<<gp, 256, 0, stream>>>(q, nullptr, Wk, bk, Kb, nullptr);
  proj_gemm_kernel<1><<<gp, 256, 0, stream>>>(q, nullptr, Wv, bv, Vb, nullptr);

  flash_attn_kernel<<<dim3(S_LEN / 32, NBATCH), 256, 0, stream>>>(Qb, Kb, Vb, Ob);

  proj_gemm_kernel<2><<<gp, 256, 0, stream>>>(nullptr, Ob, Wo, bo, nullptr, out);
}
